// MultiHeadAttention_62612033241268
// MI455X (gfx1250) — compile-verified, matrix operand order checked by reading, not run on hardware
//
#include <hip/hip_runtime.h>

typedef __attribute__((ext_vector_type(16))) __bf16 v16bf;
typedef __attribute__((ext_vector_type(8)))  __bf16 v8bf;
typedef __attribute__((ext_vector_type(8)))  float  v8f;
typedef __attribute__((ext_vector_type(2)))  float  v2f;

#define BN 32
#define TT 127
#define LL 128
#define DD 128
#define HH 8
#define HD 16

static __device__ __forceinline__ __bf16 f2bf(float f) {
  unsigned u = __builtin_bit_cast(unsigned, f);
  unsigned r = u + 0x7FFFu + ((u >> 16) & 1u);   // round-to-nearest-even
  unsigned short h = (unsigned short)(r >> 16);
  return __builtin_bit_cast(__bf16, h);
}

static __device__ __forceinline__ float gelu_f(float x) {
  return 0.5f * x * (1.0f + erff(x * 0.7071067811865476f));
}

// CDNA5 16-bit fragment (ISA 7.12.2): per lane-half, K = 8*lg + [0..7] and
// 8*lg + 16 + [0..7]  ->  exactly two contiguous 16-byte loads.
// p must already include the 8*lg element offset.
static __device__ __forceinline__ v16bf frag16(const __bf16* p) {
  v8bf lo = *(const v8bf*)(p);
  v8bf hi = *(const v8bf*)(p + 16);
  return __builtin_shufflevector(lo, hi, 0, 1, 2, 3, 4, 5, 6, 7,
                                 8, 9, 10, 11, 12, 13, 14, 15);
}

static __device__ __forceinline__ v8f wmma_bf16(v16bf a, v16bf b, v8f c) {
  return __builtin_amdgcn_wmma_f32_16x16x32_bf16(false, a, false, b, (short)0, c, false, false);
}
static __device__ __forceinline__ v8f wmma_f32(v2f a, v2f b, v8f c) {
  return __builtin_amdgcn_wmma_f32_16x16x4_f32(false, a, false, b, (short)0, c, false, false);
}

// ---------------------------------------------------------------- prepare
__global__ __launch_bounds__(256) void k_prepare(
    const float* __restrict__ padded, const int* __restrict__ masks,
    const float* __restrict__ posin, const float* __restrict__ tok,
    float* __restrict__ xg, __bf16* __restrict__ xbg,
    float* __restrict__ posg, float* __restrict__ nmask) {
  int s = blockIdx.x, tid = threadIdx.x;
  for (int idx = tid; idx < LL * DD; idx += 256) {
    int l = idx >> 7, d = idx & 127;
    float v = (l == 0) ? tok[d] : padded[((size_t)s * TT + (l - 1)) * DD + d];
    xg[(size_t)s * LL * DD + idx] = v;
    xbg[(size_t)s * LL * DD + idx] = f2bf(v);
  }
  for (int idx = tid; idx < LL * 3; idx += 256) {
    int l = idx / 3, c = idx - 3 * l;
    posg[(size_t)s * LL * 3 + idx] =
        (l == 0) ? 0.0f : posin[((size_t)s * TT + (l - 1)) * 3 + c];
  }
  for (int idx = tid; idx < LL; idx += 256) {
    int keep = (idx == 0) ? 1 : masks[(size_t)s * TT + idx - 1];
    nmask[s * LL + idx] = keep ? 0.0f : -3.402823466e38f;
  }
}

// ---------------------------------------------------------------- QKV GEMMs
__global__ __launch_bounds__(256) void k_qkv(
    const __bf16* __restrict__ xbg,
    const float* __restrict__ Wq, const float* __restrict__ bq,
    const float* __restrict__ Wk, const float* __restrict__ bk,
    const float* __restrict__ Wv, const float* __restrict__ bv,
    float* __restrict__ qg, float* __restrict__ kg, __bf16* __restrict__ vbTg) {
  __shared__ __bf16 WlT[DD * DD];   // 32 KB, stored K-contiguous: WlT[n][k]
  __shared__ float  bl[DD];
  int s = blockIdx.x, sel = blockIdx.y, tid = threadIdx.x;
  const float* W = (sel == 0) ? Wq : (sel == 1) ? Wk : Wv;
  const float* bb = (sel == 0) ? bq : (sel == 1) ? bk : bv;
  if (tid < 64) __builtin_prefetch(W + tid * 256, 0, 0);
  for (int idx = tid; idx < DD * DD; idx += 256) {
    int k = idx >> 7, n = idx & 127;
    WlT[n * DD + k] = f2bf(W[idx]);
  }
  if (tid < DD) bl[tid] = bb[tid];
  __syncthreads();
  int wid = tid >> 5, lane = tid & 31, lm = lane & 15, lg = lane >> 4;
  const __bf16* xrow = xbg + (size_t)s * LL * DD;
  int mt = wid;                       // wave = 16-row tile
  v16bf afrag[4];
  #pragma unroll
  for (int kt = 0; kt < 4; ++kt) {
    afrag[kt] = frag16(xrow + (mt * 16 + lm) * DD + kt * 32 + 8 * lg);
  }
  float* fout = (sel == 0) ? qg : kg;
  #pragma unroll 1
  for (int nt = 0; nt < 8; ++nt) {
    v8f acc = {};
    const __bf16* bp = WlT + (nt * 16 + lm) * DD + 8 * lg;
    #pragma unroll
    for (int kt = 0; kt < 4; ++kt) {
      acc = wmma_bf16(afrag[kt], frag16(bp + kt * 32), acc);
    }
    int col = nt * 16 + lm;
    float bcol = bl[col];
    if (sel == 2) {
      // vT[c][j]: rows r are contiguous -> single 16-byte store
      v8bf pack;
      #pragma unroll
      for (int r = 0; r < 8; ++r) { pack[r] = f2bf(acc[r] + bcol); }
      *(v8bf*)(vbTg + ((size_t)s * DD + col) * LL + mt * 16 + 8 * lg) = pack;
    } else {
      float* dst = fout + ((size_t)s * LL + mt * 16 + 8 * lg) * DD + col;
      #pragma unroll
      for (int r = 0; r < 8; ++r) { dst[r * DD] = acc[r] + bcol; }
    }
  }
}

// ------------------------------------------- bias MLP + logits + softmax + w@v
__global__ __launch_bounds__(256) void k_attn(
    const float* __restrict__ qg, const float* __restrict__ kg,
    const __bf16* __restrict__ vbTg, const float* __restrict__ posg,
    const float* __restrict__ nmask,
    const float* __restrict__ Wb1, const float* __restrict__ bb1,
    const float* __restrict__ Wb2, const float* __restrict__ bb2,
    float* __restrict__ wglob, float* __restrict__ attn1g) {
  __shared__ float smf[HH * 16 * LL];  // 64 KB: rel_bias f32, reused as w bf16
  int s = blockIdx.x, i0 = blockIdx.y * 16, tid = threadIdx.x;
  const float* pbase = posg + (size_t)s * LL * 3;
  // phase 1: relative positional bias MLP (VALU), 8 pairs / thread
  #pragma unroll 1
  for (int e = 0; e < 8; ++e) {
    int idx = tid + 256 * e;
    int il = idx >> 7, j = idx & 127, gi = i0 + il;
    float dx = pbase[gi * 3 + 0] - pbase[j * 3 + 0];
    float dy = pbase[gi * 3 + 1] - pbase[j * 3 + 1];
    float dz = pbase[gi * 3 + 2] - pbase[j * 3 + 2];
    float d2 = dx * dx + dy * dy + dz * dz;
    float dd = sqrtf(d2);
    float feat[6] = {dx, dy, dz, dd, dd, d2};
    float o[8];
    #pragma unroll
    for (int h = 0; h < 8; ++h) { o[h] = bb2[h]; }
    #pragma unroll 1
    for (int m = 0; m < 64; ++m) {
      float t = bb1[m];
      #pragma unroll
      for (int f = 0; f < 6; ++f) { t += feat[f] * Wb1[f * 64 + m]; }
      float g = gelu_f(t);
      #pragma unroll
      for (int h = 0; h < 8; ++h) { o[h] += g * Wb2[m * 8 + h]; }
    }
    bool zero = (gi == 0) || (j == 0);
    #pragma unroll
    for (int h = 0; h < 8; ++h) { smf[(h * 16 + il) * LL + j] = zero ? 0.0f : o[h]; }
  }
  __syncthreads();
  // phase 2: logits via fp32 WMMA (K = head dim = 16); one wave per head
  int h = tid >> 5, lane = tid & 31, lm = lane & 15, lg = lane >> 4;
  int c0 = h * HD;
  v8f t[8];
  const float* qp = qg + ((size_t)s * LL + i0) * DD + c0 + lm * DD + 2 * lg;
  v2f af[4];
  #pragma unroll
  for (int kt = 0; kt < 4; ++kt) { af[kt] = *(const v2f*)(qp + kt * 4); }
  #pragma unroll 1
  for (int nt = 0; nt < 8; ++nt) {
    v8f acc = {};
    const float* kp = kg + ((size_t)s * LL + nt * 16) * DD + c0 + lm * DD + 2 * lg;
    #pragma unroll
    for (int kt = 0; kt < 4; ++kt) {
      acc = wmma_f32(af[kt], *(const v2f*)(kp + kt * 4), acc);  // B = K^T
    }
    #pragma unroll
    for (int r = 0; r < 8; ++r) {
      int il = r + 8 * lg, j = nt * 16 + lm;
      acc[r] = acc[r] * 0.25f + smf[(h * 16 + il) * LL + j] + nmask[s * LL + j];
    }
    t[nt] = acc;
  }
  // register softmax: rows live in 16-lane half-waves -> shfl_xor 1/2/4/8
  float mx[8], sm[8];
  #pragma unroll
  for (int r = 0; r < 8; ++r) { mx[r] = -3.402823466e38f; }
  #pragma unroll
  for (int nt = 0; nt < 8; ++nt) {
    #pragma unroll
    for (int r = 0; r < 8; ++r) { mx[r] = fmaxf(mx[r], t[nt][r]); }
  }
  #pragma unroll
  for (int r = 0; r < 8; ++r) {
    #pragma unroll
    for (int m = 1; m < 16; m <<= 1) { mx[r] = fmaxf(mx[r], __shfl_xor(mx[r], m, 32)); }
  }
  #pragma unroll
  for (int r = 0; r < 8; ++r) { sm[r] = 0.0f; }
  #pragma unroll
  for (int nt = 0; nt < 8; ++nt) {
    #pragma unroll
    for (int r = 0; r < 8; ++r) {
      float e_ = __expf(t[nt][r] - mx[r]);
      t[nt][r] = e_; sm[r] += e_;
    }
  }
  #pragma unroll
  for (int r = 0; r < 8; ++r) {
    #pragma unroll
    for (int m = 1; m < 16; m <<= 1) { sm[r] += __shfl_xor(sm[r], m, 32); }
  }
  #pragma unroll
  for (int nt = 0; nt < 8; ++nt) {
    #pragma unroll
    for (int r = 0; r < 8; ++r) { t[nt][r] = t[nt][r] / sm[r]; }
  }
  __syncthreads();                       // everyone done reading bias LDS
  __bf16* wb = (__bf16*)smf;             // reuse as [8][16][128] bf16
  #pragma unroll
  for (int nt = 0; nt < 8; ++nt) {
    #pragma unroll
    for (int r = 0; r < 8; ++r) {
      int il = r + 8 * lg, j = nt * 16 + lm;
      float wv = t[nt][r];
      wb[(h * 16 + il) * LL + j] = f2bf(wv);
      wglob[(((size_t)s * HH + h) * LL + (i0 + il)) * LL + j] = wv;
    }
  }
  // attn1 = w @ v  (per head: [16x128]·[128x16], K=128 -> 4 bf16 WMMAs)
  v8f acc = {};
  const __bf16* ap = wb + (h * 16 + lm) * LL + 8 * lg;
  const __bf16* vp = vbTg + ((size_t)s * DD + c0 + lm) * LL + 8 * lg;  // vT[c][j]
  #pragma unroll
  for (int kt = 0; kt < 4; ++kt) {
    acc = wmma_bf16(frag16(ap + kt * 32), frag16(vp + kt * 32), acc);
  }
  #pragma unroll
  for (int r = 0; r < 8; ++r) {
    attn1g[((size_t)s * LL + i0 + r + 8 * lg) * DD + c0 + lm] = acc[r];
  }
}

// --------------------------- attn2: P = w_i @ G_i (WMMA), then P@Wv2 + wsum*bv2
__global__ __launch_bounds__(256) void k_attn2(
    const float* __restrict__ posg, const float* __restrict__ wglob,
    const float* __restrict__ Wv1, const float* __restrict__ bv1,
    const float* __restrict__ Wv2, const float* __restrict__ bv2,
    float* __restrict__ attn2g) {
  __shared__ __bf16 GT[DD * LL];    // 32 KB, K-contiguous: GT[m][j]
  __shared__ __bf16 wrow[16 * LL];  // 4 KB (rows 8..15 zero pad)
  __shared__ float  P[16 * DD];     // 8 KB
  __shared__ float  wsum[8];
  int s = blockIdx.x, i = blockIdx.y, tid = threadIdx.x;
  if (i == 0) {                     // rel_val row i=0 is zeroed
    if (tid < DD) attn2g[(size_t)s * LL * DD + tid] = 0.0f;
    return;
  }
  if (tid < 128) __builtin_prefetch(Wv2 + tid * DD, 0, 0);
  const float* pbase = posg + (size_t)s * LL * 3;
  // G^T: thread owns output row m (hoisted weights), contiguous j stores
  {
    int m = tid >> 1, jb = (tid & 1) * 64;
    float w0 = Wv1[0 * DD + m], w1 = Wv1[1 * DD + m], w2 = Wv1[2 * DD + m];
    float w3 = Wv1[3 * DD + m], w4 = Wv1[4 * DD + m], w5 = Wv1[5 * DD + m];
    float bb = bv1[m];
    float pix = pbase[i * 3 + 0], piy = pbase[i * 3 + 1], piz = pbase[i * 3 + 2];
    #pragma unroll 1
    for (int jj = 0; jj < 64; ++jj) {
      int j = jb + jj;
      float dx = pix - pbase[j * 3 + 0];
      float dy = piy - pbase[j * 3 + 1];
      float dz = piz - pbase[j * 3 + 2];
      float d2 = dx * dx + dy * dy + dz * dz;
      float dd = sqrtf(d2);
      float t = bb + dx * w0 + dy * w1 + dz * w2 + dd * (w3 + w4) + d2 * w5;
      GT[m * LL + j] = (j == 0) ? f2bf(0.0f) : f2bf(gelu_f(t));
    }
  }
  #pragma unroll 1
  for (int e = 0; e < 8; ++e) {
    int idx = tid + 256 * e;
    int hh = idx >> 7, jj = idx & 127;
    float wv = 0.0f;
    if (hh < 8 && jj > 0) wv = wglob[(((size_t)s * HH + hh) * LL + i) * LL + jj];
    wrow[idx] = f2bf(wv);
  }
  if (tid < 8) {
    float ssum = 0.0f;
    for (int jj = 1; jj < LL; ++jj)
      ssum += wglob[(((size_t)s * HH + tid) * LL + i) * LL + jj];
    wsum[tid] = ssum;
  }
  __syncthreads();
  int wid = tid >> 5, lane = tid & 31, lm = lane & 15, lg = lane >> 4;
  v8f acc = {};
  const __bf16* ap = wrow + lm * LL + 8 * lg;
  const __bf16* bp = GT + (wid * 16 + lm) * LL + 8 * lg;   // B = G (K=j contiguous)
  #pragma unroll
  for (int kt = 0; kt < 4; ++kt) {
    acc = wmma_bf16(frag16(ap + kt * 32), frag16(bp + kt * 32), acc);
  }
  #pragma unroll
  for (int r = 0; r < 8; ++r) { P[(r + 8 * lg) * DD + wid * 16 + lm] = acc[r]; }
  __syncthreads();
  if (tid < DD) {
    int hh = tid >> 4;
    float o = wsum[hh] * bv2[tid];
    for (int m = 0; m < DD; ++m) o += P[hh * DD + m] * Wv2[m * DD + tid];
    attn2g[((size_t)s * LL + i) * DD + tid] = o;
  }
}

// ---------------------------------------------------------------- finalize
__global__ __launch_bounds__(128) void k_finalize(
    const float* __restrict__ attn1, const float* __restrict__ attn2,
    const float* __restrict__ xg, const float* __restrict__ gamma,
    const float* __restrict__ beta, float* __restrict__ dout) {
  __shared__ float rs[4], rq[4];
  int s = blockIdx.x, d = threadIdx.x;
  size_t base = (size_t)s * LL * DD;
  float st = attn1[base + d] + attn2[base + d] + xg[base + d];
  float a = st, b = st * st;
  #pragma unroll
  for (int m = 16; m > 0; m >>= 1) {
    a += __shfl_xor(a, m, 32);
    b += __shfl_xor(b, m, 32);
  }
  if ((d & 31) == 0) { rs[d >> 5] = a; rq[d >> 5] = b; }
  __syncthreads();
  float S = rs[0] + rs[1] + rs[2] + rs[3];
  float Q = rq[0] + rq[1] + rq[2] + rq[3];
  float mu = S * (1.0f / 128.0f);
  float var = Q * (1.0f / 128.0f) - mu * mu;
  dout[(size_t)BN * TT * DD + (size_t)s * DD + d] =
      (st - mu) * rsqrtf(var + 1e-5f) * gamma[d] + beta[d];
  for (int tt = 0; tt < TT; ++tt) {
    size_t src = base + (size_t)(tt + 1) * DD + d;
    dout[((size_t)s * TT + tt) * DD + d] = attn1[src] + attn2[src] + xg[src];
  }
}

extern "C" void kernel_launch(void* const* d_in, const int* in_sizes, int n_in,
                              void* d_out, int out_size, void* d_ws, size_t ws_size,
                              hipStream_t stream) {
  (void)in_sizes; (void)n_in; (void)out_size; (void)ws_size;
  const float* padded = (const float*)d_in[0];
  const int*   masks  = (const int*)d_in[1];
  const float* posin  = (const float*)d_in[2];
  const float* Wq = (const float*)d_in[3];
  const float* bq = (const float*)d_in[4];
  const float* Wk = (const float*)d_in[5];
  const float* bk = (const float*)d_in[6];
  const float* Wv = (const float*)d_in[7];
  const float* bv = (const float*)d_in[8];
  const float* gamma = (const float*)d_in[9];
  const float* beta  = (const float*)d_in[10];
  const float* tok   = (const float*)d_in[11];
  const float* Wb1 = (const float*)d_in[12];
  const float* bb1 = (const float*)d_in[13];
  const float* Wb2 = (const float*)d_in[14];
  const float* bb2 = (const float*)d_in[15];
  const float* Wv1 = (const float*)d_in[16];
  const float* bv1 = (const float*)d_in[17];
  const float* Wv2 = (const float*)d_in[18];
  const float* bv2 = (const float*)d_in[19];

  char* ws = (char*)d_ws;
  float*  xg    = (float*)(ws + 0);                       // 2 MB
  __bf16* xbg   = (__bf16*)(ws + (2u  << 20));            // 1 MB
  float*  qg    = (float*)(ws + (3u  << 20));             // 2 MB
  float*  kg    = (float*)(ws + (5u  << 20));             // 2 MB
  __bf16* vbTg  = (__bf16*)(ws + (7u  << 20));            // 1 MB (transposed)
  float*  wglob = (float*)(ws + (8u  << 20));             // 16 MB
  float*  attn1 = (float*)(ws + (24u << 20));             // 2 MB
  float*  attn2 = (float*)(ws + (26u << 20));             // 2 MB
  float*  posg  = (float*)(ws + (28u << 20));             // 48 KB
  float*  nmask = (float*)(ws + (28u << 20) + (64u << 10));
  float*  dout  = (float*)d_out;

  k_prepare<<<BN, 256, 0, stream>>>(padded, masks, posin, tok, xg, xbg, posg, nmask);
  k_qkv<<<dim3(BN, 3), 256, 0, stream>>>(xbg, Wq, bq, Wk, bk, Wv, bv, qg, kg, vbTg);
  k_attn<<<dim3(BN, 8), 256, 0, stream>>>(qg, kg, vbTg, posg, nmask,
                                          Wb1, bb1, Wb2, bb2, wglob, attn1);
  k_attn2<<<dim3(BN, LL), 256, 0, stream>>>(posg, wglob, Wv1, bv1, Wv2, bv2, attn2);
  k_finalize<<<BN, 128, 0, stream>>>(attn1, attn2, xg, gamma, beta, dout);
}
